// NSABlock_30270929502611
// MI455X (gfx1250) — matrix operand order repaired and checked by reading, not compiled, not run on hardware
//
#include <hip/hip_runtime.h>

// ---------------------------------------------------------------------------
// NAT transformer block for MI455X (gfx1250, wave32, WMMA).
// GEMMs (QKV / proj / FC1 / FC2 = ~92% of 15.7 GFLOP) run on
// v_wmma_f32_16x16x32_bf16 with f32 accumulation; attention core (49-tap
// neighborhood, ~8% of FLOPs) is f32 VALU. HBM floor ~150-200MB @ 23.3TB/s.
// Inner loop batches all 4 B-fragment loads into one clause, then issues the
// 4 WMMAs back-to-back (single load-wait per k-tile instead of four).
// ---------------------------------------------------------------------------

#define DIMC    192
#define NHEADS  6
#define HD      32
#define KSZ     7
#define IMG     128
#define NPIX    16384        // 128*128
#define HIDDEN  768
#define LN_EPS  1e-5f

typedef __attribute__((ext_vector_type(16))) __bf16 v16bf;
typedef __attribute__((ext_vector_type(8)))  float  v8f;
typedef __attribute__((ext_vector_type(4)))  unsigned int u32x4;

union Frag16 { v16bf v; u32x4 q[2]; };

__device__ __forceinline__ unsigned short f32_to_bf16(float f) {
    unsigned int u = __float_as_uint(f);
    u += 0x7fffu + ((u >> 16) & 1u);     // round-to-nearest-even
    return (unsigned short)(u >> 16);
}

// --------------------------- fp32 -> bf16 conversion -----------------------
__global__ void cvt_bf16_kernel(const float* __restrict__ in,
                                unsigned short* __restrict__ out, int n) {
    int i = blockIdx.x * blockDim.x + threadIdx.x;
    if (i < n) out[i] = f32_to_bf16(in[i]);
}

// --------------------------- LayerNorm (wave per token) --------------------
__global__ void ln_bf16_kernel(const float* __restrict__ x,
                               const float* __restrict__ w,
                               const float* __restrict__ b,
                               unsigned short* __restrict__ out) {
    const int wave = blockIdx.x * (blockDim.x >> 5) + (threadIdx.x >> 5);
    const int lane = threadIdx.x & 31;
    const float* xp = x + (size_t)wave * DIMC;

    float vals[6];
    float s = 0.f;
    #pragma unroll
    for (int i = 0; i < 6; ++i) { vals[i] = xp[lane + 32 * i]; s += vals[i]; }
    #pragma unroll
    for (int off = 16; off > 0; off >>= 1) s += __shfl_xor(s, off, 32);
    const float mu = s * (1.f / DIMC);

    float vs = 0.f;
    #pragma unroll
    for (int i = 0; i < 6; ++i) { float d = vals[i] - mu; vs += d * d; }
    #pragma unroll
    for (int off = 16; off > 0; off >>= 1) vs += __shfl_xor(vs, off, 32);
    const float rstd = rsqrtf(vs * (1.f / DIMC) + LN_EPS);

    unsigned short* op = out + (size_t)wave * DIMC;
    #pragma unroll
    for (int i = 0; i < 6; ++i) {
        const int c = lane + 32 * i;
        op[c] = f32_to_bf16((vals[i] - mu) * rstd * w[c] + b[c]);
    }
}

// --------------------------- WMMA GEMM: Out = A @ W^T + bias ---------------
// A: [M,K] bf16 row-major.  W: [N,K] bf16 row-major (torch-style weight).
// Each wave computes a 16x64 output tile.  Per k-tile: one A fragment +
// four B fragments loaded first (one clause), then 4 back-to-back WMMAs.
// mode 0: store f32.  mode 1: out = res + |gamma[q-1]| * (acc+bias) (f32).
// mode 2: store bf16 gelu(acc+bias).
__global__ void gemm_wmma_bf16(const unsigned short* __restrict__ A,
                               const unsigned short* __restrict__ W,
                               const float* __restrict__ bias,
                               float* __restrict__ outF,
                               unsigned short* __restrict__ outH,
                               const float* __restrict__ res,
                               const float* __restrict__ gamma,   // [8, DIMC]
                               const int* __restrict__ quality,
                               int M, int N, int K, int mode) {
    const int lane = threadIdx.x & 31;
    const int ln   = lane & 15;
    const int lh   = lane >> 4;
    const int nT   = N >> 6;
    const int wid  = blockIdx.x * (blockDim.x >> 5) + (threadIdx.x >> 5);
    const int mBase = (wid / nT) << 4;
    const int nBase = (wid % nT) << 6;
    if (mBase >= M) return;                      // wave-uniform

    v8f acc[4];
    #pragma unroll
    for (int j = 0; j < 4; ++j)
        acc[j] = (v8f){0.f, 0.f, 0.f, 0.f, 0.f, 0.f, 0.f, 0.f};

    // A fragment (16x32 bf16, ISA 7.12.2 layout): lanes 0-15 hold row=lane,
    // K 0-7 (v0-3) & 16-23 (v4-7); lanes 16-31 hold K 8-15 & 24-31.
    const u32x4* ap =
        reinterpret_cast<const u32x4*>(A + (size_t)(mBase + ln) * K);
    // B fragment (32x16): lane%16 = column; lanes 0-15 K 0-7 & 16-23, lanes 16-31 K 8-15 & 24-31.
    const u32x4* wp[4];
    #pragma unroll
    for (int j = 0; j < 4; ++j) {
        const int col = nBase + (j << 4) + ln;
        wp[j] = reinterpret_cast<const u32x4*>(W + (size_t)col * K + (lh << 3));
    }

    const int kTiles = K >> 5;
    for (int kt = 0; kt < kTiles; ++kt) {
        Frag16 a;
        a.q[0] = ap[(kt << 2) + lh];
        a.q[1] = ap[(kt << 2) + 2 + lh];
        Frag16 b[4];
        #pragma unroll
        for (int j = 0; j < 4; ++j) {        // 8 b128 loads, one clause
            b[j].q[0] = wp[j][(kt << 1)];
            b[j].q[1] = wp[j][(kt << 1) + 2];
        }
        #pragma unroll
        for (int j = 0; j < 4; ++j)          // 4 WMMAs back-to-back
            acc[j] = __builtin_amdgcn_wmma_f32_16x16x32_bf16(
                false, a.v, false, b[j].v, (short)0, acc[j], false, false);
    }

    const float* grow = nullptr;
    if (mode == 1) grow = gamma + (size_t)(quality[0] - 1) * DIMC;

    // C/D layout: lanes 0-15: VGPR v -> (M=v, N=lane); lanes 16-31: M=8+v.
    #pragma unroll
    for (int j = 0; j < 4; ++j) {
        const int col  = nBase + (j << 4) + ln;
        const float bc = bias[col];
        float gc = 0.f;
        if (mode == 1) gc = fabsf(grow[col]);
        #pragma unroll
        for (int v = 0; v < 8; ++v) {
            const int row = mBase + (lh << 3) + v;
            const size_t idx = (size_t)row * N + col;
            const float val = acc[j][v] + bc;
            if (mode == 0) {
                outF[idx] = val;
            } else if (mode == 1) {
                outF[idx] = res[idx] + gc * val;
            } else {
                const float gel =
                    0.5f * val * (1.f + erff(val * 0.70710678118654752f));
                outH[idx] = f32_to_bf16(gel);
            }
        }
    }
}

// --------------------------- neighborhood attention ------------------------
// Thread per (pixel, head): q.k over clipped 7x7 window + rpb, softmax, A.v.
__global__ void na2d_attn_kernel(const float* __restrict__ qkv, // [NPIX,576]
                                 const float* __restrict__ rpb, // [6,13,13]
                                 unsigned short* __restrict__ outA) {
    const int t = blockIdx.x * blockDim.x + threadIdx.x;
    const int h = t % NHEADS;
    const int n = t / NHEADS;
    const int px = n >> 7;
    const int py = n & (IMG - 1);

    const float scale = 0.17677669529663689f;   // 32^-0.5
    float q[HD];
    const float* qp = qkv + (size_t)n * 576 + h * HD;
    #pragma unroll
    for (int d = 0; d < HD; ++d) q[d] = qp[d] * scale;

    int si = px - 3; si = si < 0 ? 0 : (si > IMG - KSZ ? IMG - KSZ : si);
    int sj = py - 3; sj = sj < 0 ? 0 : (sj > IMG - KSZ ? IMG - KSZ : sj);

    float sc[KSZ * KSZ];
    float mx = -1e30f;
    for (int ii = 0; ii < KSZ; ++ii) {
        const int ki = si + ii;
        const int ri = ki - px + (KSZ - 1);
        for (int jj = 0; jj < KSZ; ++jj) {
            const int kj = sj + jj;
            const int rj = kj - py + (KSZ - 1);
            const float* kp =
                qkv + (size_t)(ki * IMG + kj) * 576 + DIMC + h * HD;
            float s = 0.f;
            #pragma unroll
            for (int d = 0; d < HD; ++d) s += q[d] * kp[d];
            s += rpb[(h * 13 + ri) * 13 + rj];
            sc[ii * KSZ + jj] = s;
            mx = fmaxf(mx, s);
        }
    }

    float sum = 0.f;
    #pragma unroll
    for (int k = 0; k < KSZ * KSZ; ++k) { sc[k] = __expf(sc[k] - mx); sum += sc[k]; }
    const float inv = 1.f / sum;

    float o[HD];
    #pragma unroll
    for (int d = 0; d < HD; ++d) o[d] = 0.f;
    for (int ii = 0; ii < KSZ; ++ii) {
        const int ki = si + ii;
        for (int jj = 0; jj < KSZ; ++jj) {
            const int kj = sj + jj;
            const float* vp =
                qkv + (size_t)(ki * IMG + kj) * 576 + 2 * DIMC + h * HD;
            const float wgt = sc[ii * KSZ + jj] * inv;
            #pragma unroll
            for (int d = 0; d < HD; ++d) o[d] += wgt * vp[d];
        }
    }

    unsigned short* op = outA + (size_t)n * DIMC + h * HD;
    #pragma unroll
    for (int d = 0; d < HD; ++d) op[d] = f32_to_bf16(o[d]);
}

// ---------------------------------------------------------------------------
extern "C" void kernel_launch(void* const* d_in, const int* in_sizes, int n_in,
                              void* d_out, int out_size, void* d_ws, size_t ws_size,
                              hipStream_t stream) {
    const float* x      = (const float*)d_in[0];
    const float* qkv_w  = (const float*)d_in[1];
    const float* qkv_b  = (const float*)d_in[2];
    const float* proj_w = (const float*)d_in[3];
    const float* proj_b = (const float*)d_in[4];
    const float* rpb    = (const float*)d_in[5];
    const float* ln1_w  = (const float*)d_in[6];
    const float* ln1_b  = (const float*)d_in[7];
    const float* ln2_w  = (const float*)d_in[8];
    const float* ln2_b  = (const float*)d_in[9];
    const float* fc1_w  = (const float*)d_in[10];
    const float* fc1_b  = (const float*)d_in[11];
    const float* fc2_w  = (const float*)d_in[12];
    const float* fc2_b  = (const float*)d_in[13];
    const float* gamma1 = (const float*)d_in[14];
    const float* gamma2 = (const float*)d_in[15];
    const int*   quality= (const int*)d_in[16];
    float* out = (float*)d_out;

    // workspace layout (bytes, 16B-aligned chunks), ~95.3 MB total
    char* ws = (char*)d_ws;
    unsigned short* hbf    = (unsigned short*)(ws + 0);          //  6,291,456
    float*          qkvF   = (float*)         (ws + 6291456);    // 37,748,736
    unsigned short* attnbf = (unsigned short*)(ws + 44040192);   //  6,291,456
    float*          x1F    = (float*)         (ws + 50331648);   // 12,582,912
    unsigned short* h2bf   = (unsigned short*)(ws + 62914560);   //  6,291,456
    unsigned short* m1bf   = (unsigned short*)(ws + 69206016);   // 25,165,824
    unsigned short* qkvWb  = (unsigned short*)(ws + 94371840);   //    221,184
    unsigned short* projWb = (unsigned short*)(ws + 94593024);   //     73,728
    unsigned short* fc1Wb  = (unsigned short*)(ws + 94666752);   //    294,912
    unsigned short* fc2Wb  = (unsigned short*)(ws + 94961664);   //    294,912

    // 1) weights fp32 -> bf16 (tiny, L2-resident afterwards)
    cvt_bf16_kernel<<<(3 * DIMC * DIMC + 255) / 256, 256, 0, stream>>>(qkv_w, qkvWb, 3 * DIMC * DIMC);
    cvt_bf16_kernel<<<(DIMC * DIMC + 255) / 256, 256, 0, stream>>>(proj_w, projWb, DIMC * DIMC);
    cvt_bf16_kernel<<<(HIDDEN * DIMC + 255) / 256, 256, 0, stream>>>(fc1_w, fc1Wb, HIDDEN * DIMC);
    cvt_bf16_kernel<<<(DIMC * HIDDEN + 255) / 256, 256, 0, stream>>>(fc2_w, fc2Wb, DIMC * HIDDEN);

    // 2) LN1: x -> hbf (bf16)
    ln_bf16_kernel<<<NPIX / 8, 256, 0, stream>>>(x, ln1_w, ln1_b, hbf);

    // 3) QKV GEMM: [16384,192] x [576,192]^T -> qkvF (f32)
    gemm_wmma_bf16<<<(NPIX / 16) * (576 / 64) / 8, 256, 0, stream>>>(
        hbf, qkvWb, qkv_b, qkvF, nullptr, nullptr, nullptr, quality,
        NPIX, 576, DIMC, 0);

    // 4) neighborhood attention -> attnbf (bf16)
    na2d_attn_kernel<<<NPIX * NHEADS / 256, 256, 0, stream>>>(qkvF, rpb, attnbf);

    // 5) proj GEMM + residual: x1 = x + |g1| * (attn @ proj_w^T + proj_b)
    gemm_wmma_bf16<<<(NPIX / 16) * (DIMC / 64) / 8, 256, 0, stream>>>(
        attnbf, projWb, proj_b, x1F, nullptr, x, gamma1, quality,
        NPIX, DIMC, DIMC, 1);

    // 6) LN2: x1 -> h2bf
    ln_bf16_kernel<<<NPIX / 8, 256, 0, stream>>>(x1F, ln2_w, ln2_b, h2bf);

    // 7) FC1 GEMM + exact GELU -> m1bf (bf16)
    gemm_wmma_bf16<<<(NPIX / 16) * (HIDDEN / 64) / 8, 256, 0, stream>>>(
        h2bf, fc1Wb, fc1_b, nullptr, m1bf, nullptr, nullptr, quality,
        NPIX, HIDDEN, DIMC, 2);

    // 8) FC2 GEMM + residual: out = x1 + |g2| * (m @ fc2_w^T + fc2_b)
    gemm_wmma_bf16<<<(NPIX / 16) * (DIMC / 64) / 8, 256, 0, stream>>>(
        m1bf, fc2Wb, fc2_b, out, nullptr, x1F, gamma2, quality,
        NPIX, DIMC, HIDDEN, 1);
}
